// Model_16106127360570
// MI455X (gfx1250) — compile-verified, matrix operand order checked by reading, not run on hardware
//
#include <hip/hip_runtime.h>
#include <hip/hip_bf16.h>
#include <math.h>

// ---------------------------------------------------------------------------
// PVCNN forward on MI455X (gfx1250, wave32).
// GEMM work (conv3d implicit GEMM, point branch, f2) on v_wmma_f32_16x16x32_f16.
// One wave owns a 32x32 (MxN) tile: 2 M-tiles x 2 N-tiles, 4 accumulators.
// Channel-last layouts -> each fragment is 4 unconditional global_load_b128.
// K-chunks are software-pipelined with ping-pong register buffers (loads for
// chunk q+1 issue before chunk q is consumed) and amdgpu_waves_per_eu(1)
// relaxes the register-pressure target so the scheduler keeps them in flight.
// ---------------------------------------------------------------------------

typedef __attribute__((ext_vector_type(16))) _Float16 v16h;
typedef __attribute__((ext_vector_type(8)))  float    v8f;

__device__ __forceinline__ v8f wmma_f16(v16h a, v16h b, v8f c) {
  return __builtin_amdgcn_wmma_f32_16x16x32_f16(
      false, a, false, b, (short)0, c, false, false);
}

// Raw loads for a 16x32 f16 fragment row slice (ISA 7.12.2 layout):
// element e<8 -> k0+half*8+e ; e>=8 -> k0+16+half*8+(e-8).
__device__ __forceinline__ void load4(const float* __restrict__ base, unsigned off,
                                      int half, float4 q[4]) {
  const float* p = base + off + half * 8;
  q[0] = *(const float4*)(p);
  q[1] = *(const float4*)(p + 4);
  q[2] = *(const float4*)(p + 16);
  q[3] = *(const float4*)(p + 20);
}

__device__ __forceinline__ v16h cvt16(const float4 q[4]) {
  v16h f;
  f[0]  = (_Float16)q[0].x; f[1]  = (_Float16)q[0].y;
  f[2]  = (_Float16)q[0].z; f[3]  = (_Float16)q[0].w;
  f[4]  = (_Float16)q[1].x; f[5]  = (_Float16)q[1].y;
  f[6]  = (_Float16)q[1].z; f[7]  = (_Float16)q[1].w;
  f[8]  = (_Float16)q[2].x; f[9]  = (_Float16)q[2].y;
  f[10] = (_Float16)q[2].z; f[11] = (_Float16)q[2].w;
  f[12] = (_Float16)q[3].x; f[13] = (_Float16)q[3].y;
  f[14] = (_Float16)q[3].z; f[15] = (_Float16)q[3].w;
  return f;
}

__device__ __forceinline__ v16h cvt16_mask(const float4 q[4], float m) {
  v16h f;
  f[0]  = (_Float16)(q[0].x * m); f[1]  = (_Float16)(q[0].y * m);
  f[2]  = (_Float16)(q[0].z * m); f[3]  = (_Float16)(q[0].w * m);
  f[4]  = (_Float16)(q[1].x * m); f[5]  = (_Float16)(q[1].y * m);
  f[6]  = (_Float16)(q[1].z * m); f[7]  = (_Float16)(q[1].w * m);
  f[8]  = (_Float16)(q[2].x * m); f[9]  = (_Float16)(q[2].y * m);
  f[10] = (_Float16)(q[2].z * m); f[11] = (_Float16)(q[2].w * m);
  f[12] = (_Float16)(q[3].x * m); f[13] = (_Float16)(q[3].y * m);
  f[14] = (_Float16)(q[3].z * m); f[15] = (_Float16)(q[3].w * m);
  return f;
}

// ------------------------------ utility kernels ----------------------------

__global__ void zero_f32(float* __restrict__ p, size_t n) {
  size_t i = (size_t)blockIdx.x * blockDim.x + threadIdx.x;
  if (i < n) p[i] = 0.f;
}

// conv weights (Cout,Cin,27) -> tap-major padded (Cout,27,Cinp)
__global__ void pad_conv_w(const float* __restrict__ w, float* __restrict__ wp,
                           int Cin, int Cinp, int Cout) {
  int tid = blockIdx.x * blockDim.x + threadIdx.x;
  int total = Cout * 27 * Cinp;
  if (tid >= total) return;
  int ic  = tid % Cinp;
  int tap = (tid / Cinp) % 27;
  int oc  = tid / (Cinp * 27);
  wp[tid] = (ic < Cin) ? w[((size_t)oc * Cin + ic) * 27 + tap] : 0.f;
}

// dense weights (Cout,K) -> (Cout,Kp) zero-padded
__global__ void pad_point_w(const float* __restrict__ w, float* __restrict__ wp,
                            int K, int Kp, int Cout) {
  int tid = blockIdx.x * blockDim.x + threadIdx.x;
  if (tid >= Cout * Kp) return;
  int k = tid % Kp, oc = tid / Kp;
  wp[tid] = (k < K) ? w[(size_t)oc * K + k] : 0.f;
}

// (B,N,6) -> (B,N,32) zero-padded point-major copy
__global__ void transpose_in(const float* __restrict__ in, float* __restrict__ feat0,
                             int B, int N) {
  int tid = blockIdx.x * blockDim.x + threadIdx.x;
  if (tid >= B * N) return;
  const float* src = in + (size_t)tid * 6;
  float* dst = feat0 + (size_t)tid * 32;
#pragma unroll
  for (int c = 0; c < 6; ++c) dst[c] = src[c];
#pragma unroll
  for (int c = 6; c < 32; ++c) dst[c] = 0.f;
}

__global__ void coord_mean(const float* __restrict__ feat0, float* __restrict__ meanb,
                           int N) {
  __shared__ float sh[256];
  int bc = blockIdx.x;               // b*3+axis
  int b = bc / 3, axis = bc - b * 3;
  const float* src = feat0 + (size_t)b * N * 32 + axis;
  float s = 0.f;
  for (int p = threadIdx.x; p < N; p += 256) s += src[(size_t)p * 32];
  sh[threadIdx.x] = s; __syncthreads();
  for (int st = 128; st > 0; st >>= 1) {
    if (threadIdx.x < st) sh[threadIdx.x] += sh[threadIdx.x + st];
    __syncthreads();
  }
  if (threadIdx.x == 0) meanb[bc] = sh[0] / (float)N;
}

__global__ void norm_idx(const float* __restrict__ feat0, const float* __restrict__ meanb,
                         float* __restrict__ norm, int* __restrict__ idx,
                         float* __restrict__ cnt, int B, int N, int r) {
  int tid = blockIdx.x * blockDim.x + threadIdx.x;
  if (tid >= B * N) return;
  int b = tid / N, p = tid - b * N;
  int vi[3];
#pragma unroll
  for (int a = 0; a < 3; ++a) {
    float c  = feat0[((size_t)b * N + p) * 32 + a] - meanb[b * 3 + a];
    float nz = (c + 1.f) * 0.5f * (float)r;
    nz = fminf(fmaxf(nz, 0.f), (float)(r - 1));
    norm[((size_t)(b * 3 + a)) * N + p] = nz;
    vi[a] = (int)rintf(nz);               // round-half-even == jnp.round
  }
  int id = (vi[0] * r + vi[1]) * r + vi[2];
  idx[tid] = id;
  atomicAdd(&cnt[(size_t)b * r * r * r + id], 1.f);
}

// scatter-add point features into (B,R3,Cp) voxel sums, channel-fastest
__global__ void vox_scatter(const float* __restrict__ feat, int CTin, int offin,
                            const int* __restrict__ idx, float* __restrict__ sums,
                            int Cin, int Cp, int N, int R3, size_t total) {
  size_t tid = (size_t)blockIdx.x * blockDim.x + threadIdx.x;
  if (tid >= total) return;
  int c = (int)(tid % Cin);
  int p = (int)((tid / Cin) % N);
  int b = (int)(tid / ((size_t)Cin * N));
  atomicAdd(&sums[((size_t)b * R3 + idx[b * N + p]) * Cp + c],
            feat[((size_t)b * N + p) * CTin + offin + c]);
}

__global__ void grid_div(float* __restrict__ sums, const float* __restrict__ cnt,
                         int Cp, int R3, size_t total) {
  size_t tid = (size_t)blockIdx.x * blockDim.x + threadIdx.x;
  if (tid >= total) return;
  int v = (int)((tid / Cp) % R3);
  int b = (int)(tid / ((size_t)Cp * R3));
  sums[tid] = sums[tid] / fmaxf(cnt[(size_t)b * R3 + v], 1.f);
}

// --------------------- conv3d 3x3x3 as implicit GEMM (WMMA) ----------------
// gin:(B,R3,Cinp) channel-last padded  wperm:(Cout,27,Cinp)  gout:(B,R3,Cout)
// grid = (R3/32, Cout/32, B), block = 32 (one wave, 32x32 output tile).
__global__ void __launch_bounds__(32) __attribute__((amdgpu_waves_per_eu(1)))
conv3d_wmma(const float* __restrict__ gin, const float* __restrict__ wperm,
            const float* __restrict__ bias, float* __restrict__ gout,
            int Cinp, int Cout, int r, float slope) {
  const int lane = threadIdx.x;
  const int row  = lane & 15, half = lane >> 4;
  const int b    = blockIdx.z;
  const int oc0  = blockIdx.y * 32;
  const int v0   = blockIdx.x * 32;
  const int r2   = r * r, R3 = r2 * r;
  const int Kp   = 27 * Cinp;
  const float* wrow0 = wperm + (size_t)(oc0 + row) * Kp;
  const float* wrow1 = wperm + (size_t)(oc0 + 16 + row) * Kp;
  const float* ginb  = gin + (size_t)b * R3 * Cinp;

  v8f acc[2][2];                       // [nt][mt]
#pragma unroll
  for (int nt = 0; nt < 2; ++nt)
#pragma unroll
    for (int mt = 0; mt < 2; ++mt)
#pragma unroll
      for (int i = 0; i < 8; ++i) acc[nt][mt][i] = 0.f;

  int vx[2], vy[2], vz[2];
#pragma unroll
  for (int mt = 0; mt < 2; ++mt) {
    int v = v0 + mt * 16 + row;
    vz[mt] = v % r; vy[mt] = (v / r) % r; vx[mt] = v / r2;
  }

  const int cpt = Cinp >> 5;           // K-chunks per tap
  const int nch = 27 * cpt;

  auto addr = [&](int q, unsigned (&vb)[2], float (&sm)[2], unsigned& k0) {
    int tap = q / cpt;
    int ic0 = (q - tap * cpt) << 5;
    k0 = (unsigned)(tap * Cinp + ic0);
    const int dx = tap / 9 - 1, dy = (tap / 3) % 3 - 1, dz = tap % 3 - 1;
#pragma unroll
    for (int mt = 0; mt < 2; ++mt) {
      int xx = vx[mt] + dx, yy = vy[mt] + dy, zz = vz[mt] + dz;
      bool ok = ((unsigned)xx < (unsigned)r) & ((unsigned)yy < (unsigned)r) &
                ((unsigned)zz < (unsigned)r);
      int cx = min(max(xx, 0), r - 1);
      int cy = min(max(yy, 0), r - 1);
      int cz = min(max(zz, 0), r - 1);
      vb[mt] = (unsigned)((cx * r + cy) * r + cz) * (unsigned)Cinp + (unsigned)ic0;
      sm[mt] = ok ? 1.f : 0.f;
    }
  };
  auto loads = [&](unsigned k0, const unsigned (&vb)[2],
                   float4 (&wq)[2][4], float4 (&aq)[2][4]) {
    load4(wrow0, k0, half, wq[0]);
    load4(wrow1, k0, half, wq[1]);
    load4(ginb, vb[0], half, aq[0]);
    load4(ginb, vb[1], half, aq[1]);
  };
  auto compute = [&](const float4 (&wq)[2][4], const float4 (&aq)[2][4],
                     const float (&sm)[2]) {
    v16h af0 = cvt16_mask(aq[0], sm[0]);
    v16h af1 = cvt16_mask(aq[1], sm[1]);
#pragma unroll
    for (int nt = 0; nt < 2; ++nt) {
      v16h bf = cvt16(wq[nt]);
      acc[nt][0] = wmma_f16(af0, bf, acc[nt][0]);
      acc[nt][1] = wmma_f16(af1, bf, acc[nt][1]);
    }
  };

  unsigned k0A, k0B, vbA[2], vbB[2];
  float smA[2], smB[2];
  float4 wqA[2][4], wqB[2][4], aqA[2][4], aqB[2][4];

  addr(0, vbA, smA, k0A);
  loads(k0A, vbA, wqA, aqA);
  for (int q = 0; q < nch; q += 2) {
    int q1 = (q + 1 < nch) ? q + 1 : q;
    addr(q1, vbB, smB, k0B);
    loads(k0B, vbB, wqB, aqB);        // prefetch next chunk
    compute(wqA, aqA, smA);           // consume current (already resident)
    int q2 = (q + 2 < nch) ? q + 2 : q1;
    addr(q2, vbA, smA, k0A);
    loads(k0A, vbA, wqA, aqA);
    if (q + 1 < nch) compute(wqB, aqB, smB);
  }

#pragma unroll
  for (int nt = 0; nt < 2; ++nt) {
    const float bv = bias[oc0 + nt * 16 + row];            // n = lane&15
#pragma unroll
    for (int mt = 0; mt < 2; ++mt)
#pragma unroll
      for (int i = 0; i < 8; ++i) {
        int m = v0 + mt * 16 + i + 8 * half;
        float y = acc[nt][mt][i] + bv;
        y = (y > 0.f) ? y : slope * y;
        gout[((size_t)b * R3 + m) * Cout + oc0 + nt * 16 + row] = y;
      }
  }
}

// ----------------------------- squeeze-excite ------------------------------

__global__ void se_mean(const float* __restrict__ v, float* __restrict__ s0,
                        int C, int R3) {
  __shared__ float sh[256];
  int bc = blockIdx.x;               // b*C+c
  int b = bc / C, c = bc - b * C;
  const float* src = v + (size_t)b * R3 * C + c;
  float s = 0.f;
  for (int i = threadIdx.x; i < R3; i += 256) s += src[(size_t)i * C];
  sh[threadIdx.x] = s; __syncthreads();
  for (int st = 128; st > 0; st >>= 1) {
    if (threadIdx.x < st) sh[threadIdx.x] += sh[threadIdx.x + st];
    __syncthreads();
  }
  if (threadIdx.x == 0) s0[bc] = sh[0] / (float)R3;
}

__global__ void se_mlp(const float* __restrict__ s0,
                       const float* __restrict__ sw1, const float* __restrict__ sb1,
                       const float* __restrict__ sw2, const float* __restrict__ sb2,
                       float* __restrict__ s, int C) {
  __shared__ float xin[128], hid[16];
  int b = blockIdx.x, t = threadIdx.x;   // blockDim = 128
  if (t < C) xin[t] = s0[b * C + t];
  __syncthreads();
  int H = C / 8;
  if (t < H) {
    float a = sb1[t];
    for (int k = 0; k < C; ++k) a += sw1[t * C + k] * xin[k];
    hid[t] = fmaxf(a, 0.f);
  }
  __syncthreads();
  if (t < C) {
    float a = sb2[t];
    for (int k = 0; k < H; ++k) a += sw2[t * H + k] * hid[k];
    s[b * C + t] = 1.f / (1.f + __expf(-a));
  }
}

// trilinear devoxelize of (grid * s) added into (B,N,CT) concat slice
__global__ void devox_add(const float* __restrict__ grid, const float* __restrict__ norm,
                          const float* __restrict__ s, float* __restrict__ out,
                          int CTout, int offout, int C, int N, int r, size_t total) {
  size_t tid = (size_t)blockIdx.x * blockDim.x + threadIdx.x;
  if (tid >= total) return;
  int c = (int)(tid % C);
  int p = (int)((tid / C) % N);
  int b = (int)(tid / ((size_t)C * N));
  int R3 = r * r * r;
  float nx = norm[((size_t)(b * 3 + 0)) * N + p];
  float ny = norm[((size_t)(b * 3 + 1)) * N + p];
  float nz = norm[((size_t)(b * 3 + 2)) * N + p];
  int lx = (int)floorf(nx), ly = (int)floorf(ny), lz = (int)floorf(nz);
  float fx = nx - lx, fy = ny - ly, fz = nz - lz;
  const float* g = grid + (size_t)b * R3 * C + c;
  float acc = 0.f;
#pragma unroll
  for (int d = 0; d < 8; ++d) {
    int dx = (d >> 2) & 1, dy = (d >> 1) & 1, dz = d & 1;
    int cx = min(max(lx + dx, 0), r - 1);
    int cy = min(max(ly + dy, 0), r - 1);
    int cz = min(max(lz + dz, 0), r - 1);
    float wgt = (dx ? fx : 1.f - fx) * (dy ? fy : 1.f - fy) * (dz ? fz : 1.f - fz);
    acc += wgt * g[(size_t)((cx * r + cy) * r + cz) * C];
  }
  out[((size_t)b * N + p) * CTout + offout + c] += acc * s[b * C + c];
}

// ------------------- per-point 1x1 conv (WMMA), relu, store ----------------
// feat:(B,N,CTin) point-major; Kp multiple of 32; zero pad in feat/weights.
// grid = (N/32, Cout/32, B), block = 32.
__global__ void __launch_bounds__(32) __attribute__((amdgpu_waves_per_eu(1)))
pointmm_relu_wmma(const float* __restrict__ feat, int CTin, int offin,
                  const float* __restrict__ wpad, const float* __restrict__ bias,
                  float* __restrict__ out, int CTout, int offout,
                  int Kp, int N) {
  const int lane = threadIdx.x, row = lane & 15, half = lane >> 4;
  const int b = blockIdx.z, oc0 = blockIdx.y * 32, p0 = blockIdx.x * 32;
  const float* wrow0 = wpad + (size_t)(oc0 + row) * Kp;
  const float* wrow1 = wpad + (size_t)(oc0 + 16 + row) * Kp;
  const float* fb    = feat + (size_t)b * N * CTin;
  const unsigned pb0 = (unsigned)(p0 + row) * (unsigned)CTin + (unsigned)offin;
  const unsigned pb1 = (unsigned)(p0 + 16 + row) * (unsigned)CTin + (unsigned)offin;

  v8f acc[2][2];
#pragma unroll
  for (int nt = 0; nt < 2; ++nt)
#pragma unroll
    for (int mt = 0; mt < 2; ++mt)
#pragma unroll
      for (int i = 0; i < 8; ++i) acc[nt][mt][i] = 0.f;

  auto loads = [&](unsigned k0, float4 (&wq)[2][4], float4 (&aq)[2][4]) {
    load4(wrow0, k0, half, wq[0]);
    load4(wrow1, k0, half, wq[1]);
    load4(fb, pb0 + k0, half, aq[0]);
    load4(fb, pb1 + k0, half, aq[1]);
  };
  auto compute = [&](const float4 (&wq)[2][4], const float4 (&aq)[2][4]) {
    v16h af0 = cvt16(aq[0]);
    v16h af1 = cvt16(aq[1]);
#pragma unroll
    for (int nt = 0; nt < 2; ++nt) {
      v16h bf = cvt16(wq[nt]);
      acc[nt][0] = wmma_f16(af0, bf, acc[nt][0]);
      acc[nt][1] = wmma_f16(af1, bf, acc[nt][1]);
    }
  };

  const int nch = Kp >> 5;
  float4 wqA[2][4], wqB[2][4], aqA[2][4], aqB[2][4];
  loads(0, wqA, aqA);
  for (int q = 0; q < nch; q += 2) {
    unsigned k1 = (unsigned)((q + 1 < nch) ? q + 1 : q) << 5;
    loads(k1, wqB, aqB);
    compute(wqA, aqA);
    unsigned k2 = (unsigned)((q + 2 < nch) ? q + 2 : q) << 5;
    loads(k2, wqA, aqA);
    if (q + 1 < nch) compute(wqB, aqB);
  }

#pragma unroll
  for (int nt = 0; nt < 2; ++nt) {
    const float bv = bias[oc0 + nt * 16 + row];
#pragma unroll
    for (int mt = 0; mt < 2; ++mt)
#pragma unroll
      for (int i = 0; i < 8; ++i) {
        int p = p0 + mt * 16 + i + 8 * half;
        out[((size_t)b * N + p) * CTout + offout + oc0 + nt * 16 + row] =
            fmaxf(acc[nt][mt][i] + bv, 0.f);
      }
  }
}

// ---------------- f2 GEMM (WMMA) fused with global max-pool ----------------
__global__ void __launch_bounds__(32) __attribute__((amdgpu_waves_per_eu(1)))
f2max_wmma(const float* __restrict__ feat, int CTin,
           const float* __restrict__ w, const float* __restrict__ bias,
           float* __restrict__ outmax, int K, int Cout, int N) {
  const int lane = threadIdx.x, row = lane & 15, half = lane >> 4;
  const int b = blockIdx.z, oc0 = blockIdx.y * 32, p0 = blockIdx.x * 32;
  const float* wrow0 = w + (size_t)(oc0 + row) * K;
  const float* wrow1 = w + (size_t)(oc0 + 16 + row) * K;
  const float* fb    = feat + (size_t)b * N * CTin;
  const unsigned pb0 = (unsigned)(p0 + row) * (unsigned)CTin;
  const unsigned pb1 = (unsigned)(p0 + 16 + row) * (unsigned)CTin;

  v8f acc[2][2];
#pragma unroll
  for (int nt = 0; nt < 2; ++nt)
#pragma unroll
    for (int mt = 0; mt < 2; ++mt)
#pragma unroll
      for (int i = 0; i < 8; ++i) acc[nt][mt][i] = 0.f;

  auto loads = [&](unsigned k0, float4 (&wq)[2][4], float4 (&aq)[2][4]) {
    load4(wrow0, k0, half, wq[0]);
    load4(wrow1, k0, half, wq[1]);
    load4(fb, pb0 + k0, half, aq[0]);
    load4(fb, pb1 + k0, half, aq[1]);
  };
  auto compute = [&](const float4 (&wq)[2][4], const float4 (&aq)[2][4]) {
    v16h af0 = cvt16(aq[0]);
    v16h af1 = cvt16(aq[1]);
#pragma unroll
    for (int nt = 0; nt < 2; ++nt) {
      v16h bf = cvt16(wq[nt]);
      acc[nt][0] = wmma_f16(af0, bf, acc[nt][0]);
      acc[nt][1] = wmma_f16(af1, bf, acc[nt][1]);
    }
  };

  const int nch = K >> 5;
  float4 wqA[2][4], wqB[2][4], aqA[2][4], aqB[2][4];
  loads(0, wqA, aqA);
  for (int q = 0; q < nch; q += 2) {
    unsigned k1 = (unsigned)((q + 1 < nch) ? q + 1 : q) << 5;
    loads(k1, wqB, aqB);
    compute(wqA, aqA);
    unsigned k2 = (unsigned)((q + 2 < nch) ? q + 2 : q) << 5;
    loads(k2, wqA, aqA);
    if (q + 1 < nch) compute(wqB, aqB);
  }

#pragma unroll
  for (int nt = 0; nt < 2; ++nt) {
    const float bv = bias[oc0 + nt * 16 + row];
    float m = 0.f;
#pragma unroll
    for (int mt = 0; mt < 2; ++mt)
#pragma unroll
      for (int i = 0; i < 8; ++i)
        m = fmaxf(m, fmaxf(acc[nt][mt][i] + bv, 0.f));
    m = fmaxf(m, __shfl_xor(m, 16, 32));   // combine M rows 0..7 with 8..15
    if (half == 0)
      atomicMax((int*)&outmax[(size_t)b * Cout + oc0 + nt * 16 + row],
                __float_as_int(m));
  }
}

// ------------------------------- MLP head ----------------------------------
__global__ void __launch_bounds__(512)
head_kernel(const float* __restrict__ f2out,
            const float* __restrict__ h1w, const float* __restrict__ h1b,
            const float* __restrict__ h2w, const float* __restrict__ h2b,
            const float* __restrict__ h3w, const float* __restrict__ h3b,
            float* __restrict__ out) {
  __shared__ float x[1024], h1[512], h2[256], h3[40], red[2];
  int b = blockIdx.x, t = threadIdx.x;
  for (int i = t; i < 1024; i += 512) x[i] = f2out[b * 1024 + i];
  __syncthreads();
  {
    float s = h1b[t];
    for (int k = 0; k < 1024; ++k) s += h1w[t * 1024 + k] * x[k];
    h1[t] = fmaxf(s, 0.f);
  }
  __syncthreads();
  if (t < 256) {
    float s = h2b[t];
    for (int k = 0; k < 512; ++k) s += h2w[t * 512 + k] * h1[k];
    h2[t] = fmaxf(s, 0.f);
  }
  __syncthreads();
  if (t < 40) {
    float s = h3b[t];
    for (int k = 0; k < 256; ++k) s += h3w[t * 256 + k] * h2[k];
    h3[t] = s;
  }
  __syncthreads();
  if (t == 0) {
    float mx = h3[0];
    for (int i = 1; i < 40; ++i) mx = fmaxf(mx, h3[i]);
    float se = 0.f;
    for (int i = 0; i < 40; ++i) se += __expf(h3[i] - mx);
    red[0] = mx; red[1] = logf(se);
  }
  __syncthreads();
  if (t < 40) out[b * 40 + t] = h3[t] - red[0] - red[1];
}

// ---------------------------------------------------------------------------

extern "C" void kernel_launch(void* const* d_in, const int* in_sizes, int n_in,
                              void* d_out, int out_size, void* d_ws, size_t ws_size,
                              hipStream_t stream) {
  (void)in_sizes; (void)n_in; (void)out_size; (void)ws_size;
  const int B = 8, N = 8192, CT = 448;

  // JAX pytree flatten (sorted dict keys):
  // [0]=inputs; per block i (base 1+i*10): pb,pw,sb1,sb2,sw1,sw2,vb1,vb2,vw1,vw2;
  // then f2b,f2w,h1b,h1w,h2b,h2w,h3b,h3w
  const float* inputs = (const float*)d_in[0];
  auto blk = [&](int i, int j) { return (const float*)d_in[1 + i * 10 + j]; };
  const float* f2b = (const float*)d_in[51];
  const float* f2w = (const float*)d_in[52];
  const float* h1b = (const float*)d_in[53];
  const float* h1w = (const float*)d_in[54];
  const float* h2b = (const float*)d_in[55];
  const float* h2w = (const float*)d_in[56];
  const float* h3b = (const float*)d_in[57];
  const float* h3w = (const float*)d_in[58];

  // workspace carve (floats)
  float* ws = (float*)d_ws;
  size_t o = 0;
  auto alloc = [&](size_t n) { float* p = ws + o; o += n; return p; };
  float* feat0  = alloc((size_t)B * N * 32);          // (B,N,32) zero-padded
  float* meanb  = alloc((size_t)B * 3);
  float* norm32 = alloc((size_t)B * 3 * N);
  float* norm16 = alloc((size_t)B * 3 * N);
  int*   idx32  = (int*)alloc((size_t)B * N);
  int*   idx16  = (int*)alloc((size_t)B * N);
  float* cnt32  = alloc((size_t)B * 32768);
  float* cnt16  = alloc((size_t)B * 4096);
  float* concat = alloc((size_t)B * N * CT);          // (B,N,448)
  float* bufA   = alloc((size_t)B * 32768 * 128);     // (B,R3,Cp) grids
  float* bufB   = alloc((size_t)B * 32768 * 128);
  float* f2out  = alloc((size_t)B * 1024);
  float* se_s0  = alloc((size_t)B * 128);
  float* se_s   = alloc((size_t)B * 128);
  float* wpadC1 = alloc((size_t)128 * 27 * 128);      // conv1 permuted+padded
  float* wpadC2 = alloc((size_t)128 * 27 * 128);      // conv2 permuted+padded
  float* wpadP  = alloc((size_t)128 * 128);           // point weights padded

  auto nb = [](size_t n) { return (unsigned)((n + 255) / 256); };

  // prep
  transpose_in<<<nb((size_t)B * N), 256, 0, stream>>>(inputs, feat0, B, N);
  coord_mean<<<B * 3, 256, 0, stream>>>(feat0, meanb, N);
  zero_f32<<<nb((size_t)B * 32768), 256, 0, stream>>>(cnt32, (size_t)B * 32768);
  zero_f32<<<nb((size_t)B * 4096), 256, 0, stream>>>(cnt16, (size_t)B * 4096);
  norm_idx<<<nb((size_t)B * N), 256, 0, stream>>>(feat0, meanb, norm32, idx32, cnt32, B, N, 32);
  norm_idx<<<nb((size_t)B * N), 256, 0, stream>>>(feat0, meanb, norm16, idx16, cnt16, B, N, 16);

  const int cins[5]  = {6, 64, 128, 64, 128};
  const int couts[5] = {64, 128, 64, 128, 64};
  const int rs[5]    = {32, 16, 32, 16, 32};
  const int offs[5]  = {0, 64, 192, 256, 384};

  const float* feat = feat0;
  int CTin = 32, offin = 0;
  for (int i = 0; i < 5; ++i) {
    const int Cin = cins[i], Cout = couts[i], r = rs[i], R3 = r * r * r;
    const int Cinp1 = (Cin + 31) / 32 * 32;   // conv1 channel pad
    const int Cinp2 = Cout;                   // conv2 input: already 32x
    const int KpP   = (Cin + 31) / 32 * 32;   // point branch K pad
    const float* normp = (r == 32) ? norm32 : norm16;
    const int*   idxp  = (r == 32) ? idx32  : idx16;
    const float* cntp  = (r == 32) ? cnt32  : cnt16;

    // pad/permute weights for this block
    pad_conv_w<<<nb((size_t)Cout * 27 * Cinp1), 256, 0, stream>>>(blk(i, 8), wpadC1, Cin,  Cinp1, Cout);
    pad_conv_w<<<nb((size_t)Cout * 27 * Cinp2), 256, 0, stream>>>(blk(i, 9), wpadC2, Cout, Cinp2, Cout);
    pad_point_w<<<nb((size_t)Cout * KpP), 256, 0, stream>>>(blk(i, 1), wpadP, Cin, KpP, Cout);

    // voxelize: scatter-mean into bufA (B,R3,Cinp1), channel-last
    size_t gsz = (size_t)B * R3 * Cinp1;
    zero_f32<<<nb(gsz), 256, 0, stream>>>(bufA, gsz);
    vox_scatter<<<nb((size_t)B * Cin * N), 256, 0, stream>>>(
        feat, CTin, offin, idxp, bufA, Cin, Cinp1, N, R3, (size_t)B * Cin * N);
    grid_div<<<nb(gsz), 256, 0, stream>>>(bufA, cntp, Cinp1, R3, gsz);

    // conv1 (bufA -> bufB), conv2 (bufB -> bufA), both leaky 0.1
    conv3d_wmma<<<dim3(R3 / 32, Cout / 32, B), 32, 0, stream>>>(
        bufA, wpadC1, blk(i, 6), bufB, Cinp1, Cout, r, 0.1f);
    conv3d_wmma<<<dim3(R3 / 32, Cout / 32, B), 32, 0, stream>>>(
        bufB, wpadC2, blk(i, 7), bufA, Cinp2, Cout, r, 0.1f);

    // squeeze-excite
    se_mean<<<B * Cout, 256, 0, stream>>>(bufA, se_s0, Cout, R3);
    se_mlp<<<B, 128, 0, stream>>>(se_s0, blk(i, 4), blk(i, 2), blk(i, 5), blk(i, 3),
                                  se_s, Cout);

    // point branch writes relu into concat slice, devox adds on top
    pointmm_relu_wmma<<<dim3(N / 32, Cout / 32, B), 32, 0, stream>>>(
        feat, CTin, offin, wpadP, blk(i, 0), concat, CT, offs[i], KpP, N);
    devox_add<<<nb((size_t)B * Cout * N), 256, 0, stream>>>(
        bufA, normp, se_s, concat, CT, offs[i], Cout, N, r, (size_t)B * Cout * N);

    feat = concat; CTin = CT; offin = offs[i];
  }

  // f2 (448 -> 1024) fused with max-pool over points (K=448 = 14*32 exact)
  zero_f32<<<nb((size_t)B * 1024), 256, 0, stream>>>(f2out, (size_t)B * 1024);
  f2max_wmma<<<dim3(N / 32, 1024 / 32, B), 32, 0, stream>>>(
      concat, CT, f2w, f2b, f2out, CT, 1024, N);

  // MLP head + log_softmax
  head_kernel<<<B, 512, 0, stream>>>(f2out, h1w, h1b, h2w, h2b, h3w, h3b,
                                     (float*)d_out);
}
